// BasicTransformerBlock_75436805587448
// MI455X (gfx1250) — matrix operand order repaired and checked by reading, not compiled, not run on hardware
//
#include <hip/hip_runtime.h>
#include <hip/hip_bf16.h>

// ---- problem constants (match reference) ----
#define TD    1024        // token dim D
#define NH    16          // heads
#define DHD   64          // head dim
#define DIFF  4096        // FFN inner
#define TSEQ  1024        // T
#define NB    4           // batch
#define NTOK  (NB*TSEQ)   // 4096
#define MAXRELC 32

typedef __attribute__((ext_vector_type(16))) _Float16 v16h;
typedef __attribute__((ext_vector_type(8)))  float    v8f;
typedef __attribute__((ext_vector_type(4)))  _Float16 h4;
typedef int v4i __attribute__((vector_size(16)));

union FragH { v16h v; _Float16 h[16]; };

__device__ __forceinline__ v8f wmma16(v16h a, v16h b, v8f c) {
  // D = A(16x32 f16) * B(32x16 f16) + C(16x16 f32)
  return __builtin_amdgcn_wmma_f32_16x16x32_f16(false, a, false, b, (short)0, c,
                                                false, false);
}

// A-fragment K index for element e of the v16h, given lane-half (ISA 7.12.2):
// VGPR j (j=0..3): lanes0-15 K=2j,2j+1 ; lanes16-31 K=2j+8,2j+9
// VGPR 4+j:        K = 16+2j(+1)       ; K = 24+2j(+1)
__device__ __forceinline__ int akk(int e, int half) {
  int j = e >> 1;
  return ((j & 4) << 2) + ((j & 3) << 1) + (half << 3) + (e & 1);
}

// ---- CDNA5 async global->LDS path (ASYNCcnt), with vector-copy fallback ----
#if defined(__HIP_DEVICE_COMPILE__) && defined(__has_builtin)
#  if __has_builtin(__builtin_amdgcn_global_load_async_to_lds_b128) && \
      __has_builtin(__builtin_amdgcn_s_wait_asynccnt)
#    define USE_ASYNC 1
#  endif
#endif
#ifndef USE_ASYNC
#  define USE_ASYNC 0
#endif

__device__ __forceinline__ void copy16(void* lds_dst, const void* gsrc) {
#if USE_ASYNC
  __builtin_amdgcn_global_load_async_to_lds_b128(
      (__attribute__((address_space(1))) v4i*)gsrc,
      (__attribute__((address_space(3))) v4i*)(unsigned int)(size_t)lds_dst,
      0, 0);
#else
  *(uint4*)lds_dst = *(const uint4*)gsrc;
#endif
}
__device__ __forceinline__ void copy_fence() {
#if USE_ASYNC
  __builtin_amdgcn_s_wait_asynccnt(0);
#endif
  __syncthreads();
}

// ====================== tiled WMMA GEMM:  C = A * W^T (+bias)(+resid) =========
// A: (M,K) f16 row-major ; W: (N,K) f16 row-major (so W^T columns are contiguous)
#define GBM 128
#define GBN 64
#define GBK 32

__global__ __launch_bounds__(256)
void k_gemm(const _Float16* __restrict__ A, const _Float16* __restrict__ W,
            const float* __restrict__ bias, const float* __restrict__ resid,
            float* __restrict__ outf, _Float16* __restrict__ outh,
            int M, int N, int K)
{
  __shared__ __align__(16) _Float16 As[GBM][GBK];
  __shared__ __align__(16) _Float16 Bs[GBN][GBK];
  uint4* AsV = (uint4*)&As[0][0];   // 512 x uint4
  uint4* BsV = (uint4*)&Bs[0][0];   // 256 x uint4

  const int tid  = threadIdx.x;
  const int lane = tid & 31;
  const int wid  = tid >> 5;      // 8 waves
  const int wm   = wid & 3;       // 4 row groups of 32
  const int wn   = wid >> 2;      // 2 col groups of 32
  const int m16  = lane & 15;
  const int half = lane >> 4;

  const int m0 = blockIdx.y * GBM;
  const int n0 = blockIdx.x * GBN;
  (void)M;

  const size_t K8 = (size_t)K >> 3;                  // row stride in uint4
  const uint4* Ag = (const uint4*)A + (size_t)m0 * K8;
  const uint4* Wg = (const uint4*)W + (size_t)n0 * K8;
  const int tr = tid >> 2, tc = tid & 3;             // per-thread tile coords

  v8f acc[2][2] = {};

  for (int k0 = 0; k0 < K; k0 += GBK) {
    const size_t kb8 = (size_t)k0 >> 3;
    copy16(&AsV[tid],       &Ag[(size_t)tr * K8 + kb8 + tc]);
    copy16(&AsV[tid + 256], &Ag[(size_t)(tr + 64) * K8 + kb8 + tc]);
    copy16(&BsV[tid],       &Wg[(size_t)tr * K8 + kb8 + tc]);
    if (k0 + GBK < K) {   // CDNA5 global_prefetch of next tiles
      __builtin_prefetch(&Ag[(size_t)tr * K8 + kb8 + 4 + tc], 0, 1);
      __builtin_prefetch(&Wg[(size_t)tr * K8 + kb8 + 4 + tc], 0, 1);
    }
    copy_fence();

    FragH af[2], bf[2];
#pragma unroll
    for (int ti = 0; ti < 2; ++ti) {
      const int row = wm * 32 + ti * 16 + m16;
#pragma unroll
      for (int e = 0; e < 16; ++e) af[ti].h[e] = As[row][half * 16 + e];
    }
#pragma unroll
    for (int tj = 0; tj < 2; ++tj) {
      const int col = wn * 32 + tj * 16 + m16;   // B-frag lane n = col
      const int kb  = half * 16;                 // 16 contiguous K per half
#pragma unroll
      for (int e = 0; e < 16; ++e) bf[tj].h[e] = Bs[col][kb + e];
    }
#pragma unroll
    for (int ti = 0; ti < 2; ++ti)
#pragma unroll
      for (int tj = 0; tj < 2; ++tj)
        acc[ti][tj] = wmma16(af[ti].v, bf[tj].v, acc[ti][tj]);
    __syncthreads();
  }

  // epilogue; C/D layout: lane col = L%16, rows r + 8*(L/16)
#pragma unroll
  for (int ti = 0; ti < 2; ++ti)
#pragma unroll
    for (int tj = 0; tj < 2; ++tj)
#pragma unroll
      for (int r = 0; r < 8; ++r) {
        const int row = m0 + wm * 32 + ti * 16 + r + 8 * half;
        const int col = n0 + wn * 32 + tj * 16 + m16;
        float v = acc[ti][tj][r];
        if (bias)  v += bias[col];
        if (resid) v += resid[(size_t)row * N + col];
        if (outf) outf[(size_t)row * N + col] = v;
        else      outh[(size_t)row * N + col] = (_Float16)v;
      }
}

// ====================== flash attention (one (b,h), 64 q-rows per block) ======
__global__ __launch_bounds__(128)
void k_attn(const _Float16* __restrict__ Q, const _Float16* __restrict__ Kv,
            const _Float16* __restrict__ Vv, const float* __restrict__ rel,
            _Float16* __restrict__ Out)
{
  __shared__ __align__(16) _Float16 Qs[64][DHD];     // 8 KB
  __shared__ __align__(16) _Float16 Ks[32][DHD];     // K chunk [tk][dh]
  __shared__ __align__(16) _Float16 Vs[32][DHD];     // V chunk [tk][dh]
  __shared__ __align__(16) _Float16 Ps[4][16][32];   // per-wave P relayout
  __shared__ float rel_s[2 * MAXRELC + 1];

  const int tid  = threadIdx.x;
  const int lane = tid & 31;
  const int wid  = tid >> 5;       // 4 waves, 16 q-rows each
  const int m16  = lane & 15;
  const int half = lane >> 4;

  const int b  = blockIdx.y >> 4;
  const int h  = blockIdx.y & 15;
  const int q0 = blockIdx.x * 64;

  if (rel)
    for (int i = tid; i < 2 * MAXRELC + 1; i += 128)
      rel_s[i] = rel[h * (2 * MAXRELC + 1) + i];

  // stage Q tile (64 rows x 64 dh) in LDS, b128 granularity
  {
    const uint4* Qg = (const uint4*)Q;
    uint4* QsV = (uint4*)&Qs[0][0];          // 512 x uint4
#pragma unroll
    for (int p = 0; p < 4; ++p) {
      const int i = tid + p * 128;
      const int r = i >> 3, c = i & 7;       // 8 uint4 per row
      copy16(&QsV[i],
             &Qg[((((size_t)(b * TSEQ + q0 + r)) * TD + h * DHD) >> 3) + c]);
    }
  }
  copy_fence();

  // Q fragments: 16 rows x 64 dh as two 16x32 A-fragments (from LDS)
  FragH qf[2];
#pragma unroll
  for (int s = 0; s < 2; ++s)
#pragma unroll
    for (int e = 0; e < 16; ++e)
      qf[s].h[e] = Qs[wid * 16 + m16][s * 32 + half * 16 + e];

  float m_run[8], l_run[8], scl[8];
#pragma unroll
  for (int r = 0; r < 8; ++r) { m_run[r] = -1e30f; l_run[r] = 0.f; }
  v8f o[4] = {};

  for (int tk0 = 0; tk0 < TSEQ; tk0 += 32) {
    {
      const uint4* Kg = (const uint4*)Kv;
      const uint4* Vg = (const uint4*)Vv;
      uint4* KsV = (uint4*)&Ks[0][0];        // 256 x uint4
      uint4* VsV = (uint4*)&Vs[0][0];
#pragma unroll
      for (int p = 0; p < 2; ++p) {
        const int i = tid + p * 128;
        const int r = i >> 3, c = i & 7;
        const size_t g = ((((size_t)(b * TSEQ + tk0 + r)) * TD + h * DHD) >> 3) + c;
        copy16(&KsV[i], &Kg[g]);
        copy16(&VsV[i], &Vg[g]);
      }
    }
    copy_fence();

    // scores S(16x32) = Q(16x64) * K^T(64x32): 2 n-tiles x 2 k-slices
    FragH kf[2][2];
#pragma unroll
    for (int nt = 0; nt < 2; ++nt)
#pragma unroll
      for (int s = 0; s < 2; ++s)
#pragma unroll
        for (int e = 0; e < 16; ++e)
          kf[nt][s].h[e] = Ks[nt * 16 + m16][s * 32 + half * 16 + e];

    v8f S0 = {}, S1 = {};
    S0 = wmma16(qf[0].v, kf[0][0].v, S0);
    S0 = wmma16(qf[1].v, kf[0][1].v, S0);
    S1 = wmma16(qf[0].v, kf[1][0].v, S1);
    S1 = wmma16(qf[1].v, kf[1][1].v, S1);

    // online softmax over this 32-wide chunk
#pragma unroll
    for (int r = 0; r < 8; ++r) {
      float s0 = S0[r] * 0.125f;    // DH^-0.5
      float s1 = S1[r] * 0.125f;
      if (rel) {
        const int row = q0 + wid * 16 + r + 8 * half;
        int d0 = row - (tk0 + m16);
        int d1 = row - (tk0 + 16 + m16);
        d0 = min(max(d0, -MAXRELC), MAXRELC) + MAXRELC;
        d1 = min(max(d1, -MAXRELC), MAXRELC) + MAXRELC;
        s0 += rel_s[d0];
        s1 += rel_s[d1];
      }
      float cm = fmaxf(s0, s1);
#pragma unroll
      for (int off = 1; off < 16; off <<= 1)
        cm = fmaxf(cm, __shfl_xor(cm, off, 32));
      const float m_new = fmaxf(m_run[r], cm);
      const float sc = __expf(m_run[r] - m_new);
      const float p0 = __expf(s0 - m_new);
      const float p1 = __expf(s1 - m_new);
      float rs = p0 + p1;
#pragma unroll
      for (int off = 1; off < 16; off <<= 1)
        rs += __shfl_xor(rs, off, 32);
      l_run[r] = l_run[r] * sc + rs;
      m_run[r] = m_new;
      scl[r] = sc;
      const int rowloc = r + 8 * half;
      Ps[wid][rowloc][m16]      = (_Float16)p0;
      Ps[wid][rowloc][16 + m16] = (_Float16)p1;
    }
#pragma unroll
    for (int t = 0; t < 4; ++t)
#pragma unroll
      for (int r = 0; r < 8; ++r)
        o[t][r] *= scl[r];

    __syncthreads();

    // O(16x64) += P(16x32) * V(32x64): A-frag from LDS, 4 B n-tiles
    FragH pf;
#pragma unroll
    for (int e = 0; e < 16; ++e) pf.h[e] = Ps[wid][m16][half * 16 + e];
#pragma unroll
    for (int tj = 0; tj < 4; ++tj) {
      FragH vf;
#pragma unroll
      for (int e = 0; e < 16; ++e)
        vf.h[e] = Vs[half * 16 + e][tj * 16 + m16];   // B[k][n] = V[tk][dh]
      o[tj] = wmma16(pf.v, vf.v, o[tj]);
    }
    __syncthreads();
  }

#pragma unroll
  for (int tj = 0; tj < 4; ++tj)
#pragma unroll
    for (int r = 0; r < 8; ++r) {
      const int row = q0 + wid * 16 + r + 8 * half;
      const int dh  = tj * 16 + m16;
      Out[((size_t)(b * TSEQ + row)) * TD + h * DHD + dh] =
          (_Float16)(o[tj][r] / l_run[r]);
    }
}

// ====================== equivariant LN (+ optional AdaLN scale/shift) =========
__global__ __launch_bounds__(256)
void k_adaln(const float* __restrict__ x, const float* __restrict__ ss,
             _Float16* __restrict__ out)
{
  __shared__ float sb[256];
  const int c   = threadIdx.x;
  const int row = blockIdx.x >> 2;
  const int g   = blockIdx.x & 3;
  const int Dg  = TD / 4;   // 256
  const float v = x[(size_t)row * TD + g * Dg + c];

  sb[c] = v; __syncthreads();
  for (int s = 128; s > 0; s >>= 1) { if (c < s) sb[c] += sb[c + s]; __syncthreads(); }
  const float mu = sb[0] / (float)Dg;
  __syncthreads();
  sb[c] = v * v; __syncthreads();
  for (int s = 128; s > 0; s >>= 1) { if (c < s) sb[c] += sb[c + s]; __syncthreads(); }
  const float var = sb[0] / (float)Dg - mu * mu;

  float n = (v - mu) * rsqrtf(var + 1e-5f);
  const int d = g * Dg + c;
  if (ss)
    n = n * (1.f + ss[(size_t)row * (2 * TD) + d]) + ss[(size_t)row * (2 * TD) + TD + d];
  out[(size_t)row * TD + d] = (_Float16)n;
}

// ====================== elementwise helpers (4-wide) ==========================
__global__ void k_silu(const float4* __restrict__ in, h4* __restrict__ out, int n4) {
  int i = blockIdx.x * 256 + threadIdx.x;
  if (i < n4) {
    float4 f = in[i];
    h4 o = { (_Float16)(f.x / (1.f + __expf(-f.x))),
             (_Float16)(f.y / (1.f + __expf(-f.y))),
             (_Float16)(f.z / (1.f + __expf(-f.z))),
             (_Float16)(f.w / (1.f + __expf(-f.w))) };
    out[i] = o;
  }
}
__global__ void k_cvt(const float4* __restrict__ in, h4* __restrict__ out, int n4) {
  int i = blockIdx.x * 256 + threadIdx.x;
  if (i < n4) {
    float4 f = in[i];
    h4 o = { (_Float16)f.x, (_Float16)f.y, (_Float16)f.z, (_Float16)f.w };
    out[i] = o;
  }
}
__device__ __forceinline__ float gelu_e(float g) {
  return 0.5f * g * (1.f + erff(g * 0.70710678118f));
}
__global__ void k_geglu(const float4* __restrict__ h12, h4* __restrict__ out) {
  int i = blockIdx.x * 256 + threadIdx.x;       // NTOK*DIFF/4 total, exact grid
  int row = i >> 10, jj = i & 1023;             // 1024 float4 per DIFF row
  float4 a = h12[(size_t)row * 2048 + jj];          // 2*DIFF/4 per row
  float4 g = h12[(size_t)row * 2048 + 1024 + jj];
  h4 o = { (_Float16)(a.x * gelu_e(g.x)), (_Float16)(a.y * gelu_e(g.y)),
           (_Float16)(a.z * gelu_e(g.z)), (_Float16)(a.w * gelu_e(g.w)) };
  out[i] = o;
}

// ====================== host orchestration ====================================
extern "C" void kernel_launch(void* const* d_in, const int* in_sizes, int n_in,
                              void* d_out, int out_size, void* d_ws, size_t ws_size,
                              hipStream_t stream)
{
  (void)in_sizes; (void)n_in; (void)out_size; (void)ws_size;
  const float* hidden = (const float*)d_in[0];
  const float* enc    = (const float*)d_in[1];
  const float* temb   = (const float*)d_in[2];
  const float* w_ada1 = (const float*)d_in[3];
  const float* b_ada1 = (const float*)d_in[4];
  const float* w_ada2 = (const float*)d_in[5];
  const float* b_ada2 = (const float*)d_in[6];
  const float* wq1 = (const float*)d_in[7];
  const float* wk1 = (const float*)d_in[8];
  const float* wv1 = (const float*)d_in[9];
  const float* wo1 = (const float*)d_in[10];
  const float* bo1 = (const float*)d_in[11];
  const float* rel = (const float*)d_in[12];
  const float* wq2 = (const float*)d_in[13];
  const float* wk2 = (const float*)d_in[14];
  const float* wv2 = (const float*)d_in[15];
  const float* wo2 = (const float*)d_in[16];
  const float* bo2 = (const float*)d_in[17];
  const float* w_ff1 = (const float*)d_in[18];
  const float* b_ff1 = (const float*)d_in[19];
  const float* w_ff2 = (const float*)d_in[20];
  const float* b_ff2 = (const float*)d_in[21];

  char* p = (char*)d_ws;
  auto carve = [&](size_t bytes) -> void* {
    void* q = (void*)p; p += (bytes + 255) & ~(size_t)255; return q;
  };
  const size_t MD = (size_t)NTOK * TD;        // 4M
  // f16 scratch
  _Float16* wada1h = (_Float16*)carve((size_t)2 * TD * TD * 2);
  _Float16* wada2h = (_Float16*)carve((size_t)2 * TD * TD * 2);
  _Float16* wq1h = (_Float16*)carve((size_t)TD * TD * 2);
  _Float16* wk1h = (_Float16*)carve((size_t)TD * TD * 2);
  _Float16* wv1h = (_Float16*)carve((size_t)TD * TD * 2);
  _Float16* wo1h = (_Float16*)carve((size_t)TD * TD * 2);
  _Float16* wq2h = (_Float16*)carve((size_t)TD * TD * 2);
  _Float16* wk2h = (_Float16*)carve((size_t)TD * TD * 2);
  _Float16* wv2h = (_Float16*)carve((size_t)TD * TD * 2);
  _Float16* wo2h = (_Float16*)carve((size_t)TD * TD * 2);
  _Float16* wff1h = (_Float16*)carve((size_t)2 * DIFF * TD * 2);
  _Float16* wff2h = (_Float16*)carve((size_t)TD * DIFF * 2);
  _Float16* sh    = (_Float16*)carve(MD * 2);  // silu(temb)
  _Float16* ench  = (_Float16*)carve(MD * 2);
  _Float16* x1h   = (_Float16*)carve(MD * 2);  // normalized activations
  _Float16* qh    = (_Float16*)carve(MD * 2);
  _Float16* kh    = (_Float16*)carve(MD * 2);
  _Float16* vh    = (_Float16*)carve(MD * 2);
  _Float16* ah    = (_Float16*)carve(MD * 2);  // attention out
  _Float16* ggh   = (_Float16*)carve((size_t)NTOK * DIFF * 2);
  // f32 scratch
  float* x   = (float*)carve(MD * 4);
  float* ss  = (float*)carve((size_t)NTOK * 2 * TD * 4);
  float* h12 = (float*)carve((size_t)NTOK * 2 * DIFF * 4);

  auto CVT = [&](const float* src, _Float16* dst, size_t n) {
    k_cvt<<<dim3((unsigned)(n / 1024)), dim3(256), 0, stream>>>(
        (const float4*)src, (h4*)dst, (int)(n / 4));
  };
  auto GEMM = [&](const _Float16* A, const _Float16* W, const float* bias,
                  const float* resid, float* outf, _Float16* outh,
                  int M, int Nn, int K) {
    dim3 grid(Nn / GBN, M / GBM);
    k_gemm<<<grid, dim3(256), 0, stream>>>(A, W, bias, resid, outf, outh, M, Nn, K);
  };

  // weight / input conversions (fp32 -> f16)
  CVT(w_ada1, wada1h, (size_t)2 * TD * TD);
  CVT(w_ada2, wada2h, (size_t)2 * TD * TD);
  CVT(wq1, wq1h, (size_t)TD * TD);  CVT(wk1, wk1h, (size_t)TD * TD);
  CVT(wv1, wv1h, (size_t)TD * TD);  CVT(wo1, wo1h, (size_t)TD * TD);
  CVT(wq2, wq2h, (size_t)TD * TD);  CVT(wk2, wk2h, (size_t)TD * TD);
  CVT(wv2, wv2h, (size_t)TD * TD);  CVT(wo2, wo2h, (size_t)TD * TD);
  CVT(w_ff1, wff1h, (size_t)2 * DIFF * TD);
  CVT(w_ff2, wff2h, (size_t)TD * DIFF);
  CVT(enc, ench, MD);
  k_silu<<<dim3((unsigned)(MD / 1024)), dim3(256), 0, stream>>>(
      (const float4*)temb, (h4*)sh, (int)(MD / 4));

  // running hidden state x = hidden_states
  (void)hipMemcpyAsync(x, hidden, MD * 4, hipMemcpyDeviceToDevice, stream);

  const dim3 lnGrid(NTOK * 4), lnBlk(256);
  const dim3 atGrid(TSEQ / 64, NB * NH), atBlk(128);

  // ---- 1) AdaLN1 + self-attention (+rel bias) + residual ----
  GEMM(sh, wada1h, b_ada1, nullptr, ss, nullptr, NTOK, 2 * TD, TD);
  k_adaln<<<lnGrid, lnBlk, 0, stream>>>(x, ss, x1h);
  GEMM(x1h, wq1h, nullptr, nullptr, nullptr, qh, NTOK, TD, TD);
  GEMM(x1h, wk1h, nullptr, nullptr, nullptr, kh, NTOK, TD, TD);
  GEMM(x1h, wv1h, nullptr, nullptr, nullptr, vh, NTOK, TD, TD);
  k_attn<<<atGrid, atBlk, 0, stream>>>(qh, kh, vh, rel, ah);
  GEMM(ah, wo1h, bo1, x, x, nullptr, NTOK, TD, TD);          // x += proj(attn)

  // ---- 2) AdaLN2 + cross-attention + residual ----
  GEMM(sh, wada2h, b_ada2, nullptr, ss, nullptr, NTOK, 2 * TD, TD);
  k_adaln<<<lnGrid, lnBlk, 0, stream>>>(x, ss, x1h);
  GEMM(x1h, wq2h, nullptr, nullptr, nullptr, qh, NTOK, TD, TD);
  GEMM(ench, wk2h, nullptr, nullptr, nullptr, kh, NTOK, TD, TD);
  GEMM(ench, wv2h, nullptr, nullptr, nullptr, vh, NTOK, TD, TD);
  k_attn<<<atGrid, atBlk, 0, stream>>>(qh, kh, vh, nullptr, ah);
  GEMM(ah, wo2h, bo2, x, x, nullptr, NTOK, TD, TD);          // x += proj(attn)

  // ---- 3) eq_ln + GEGLU FFN + residual -> d_out ----
  k_adaln<<<lnGrid, lnBlk, 0, stream>>>(x, nullptr, x1h);
  GEMM(x1h, wff1h, b_ff1, nullptr, h12, nullptr, NTOK, 2 * DIFF, TD);
  k_geglu<<<dim3((unsigned)((size_t)NTOK * DIFF / 1024)), dim3(256), 0, stream>>>(
      (const float4*)h12, (h4*)ggh);
  GEMM(ggh, wff2h, b_ff2, x, (float*)d_out, nullptr, NTOK, TD, DIFF);
}
